// MultiHeadAttention_11106785427960
// MI455X (gfx1250) — matrix operand order repaired and checked by reading, not compiled, not run on hardware
//
#include <hip/hip_runtime.h>
#include <hip/hip_bf16.h>
#include <stdint.h>

// Problem constants (match reference)
#define BB 2
#define SS 2048
#define DDIM 1024
#define HH 16
#define HD 64

typedef __attribute__((ext_vector_type(16))) __bf16 v16bf;
typedef __attribute__((ext_vector_type(8)))  float  v8f;
typedef __attribute__((ext_vector_type(4)))  unsigned int v4u;
typedef __attribute__((ext_vector_type(8)))  int v8i;
typedef __attribute__((ext_vector_type(4)))  int v4i;

union BfFrag { v16bf v; unsigned short u[16]; };

__device__ __forceinline__ unsigned short f2bf(float f) {
  union { float f; uint32_t u; } c; c.f = f;
  uint32_t r = c.u + 0x7fffu + ((c.u >> 16) & 1u);   // round-to-nearest-even
  return (unsigned short)(r >> 16);
}

// A-matrix (16-bit, 16x32) element->K mapping, per ISA 7.12.2.
// Per lane this yields two contiguous 8-half runs -> ds_load_b128 pairs.
__device__ __forceinline__ int aKidx(int e, int lane) {
  int r = e >> 1, hi = e & 1;
  return 2 * (r & 3) + hi + ((r >= 4) ? 16 : 0) + ((lane & 16) ? 8 : 0);
}
// B-matrix (16-bit, 32x16): same K order as A (lanes 0-15: K=0..7,16..23; lanes 16-31: K=8..15,24..31).
// bKidx(e,lane) == aKidx(e,lane).
__device__ __forceinline__ int bKidx(int e, int lane) {
  return (e & 7) + ((e >= 8) ? 16 : 0) + ((lane & 16) ? 8 : 0);
}

__device__ __forceinline__ v8f wmma_bf16(v16bf a, v16bf b, v8f c) {
  return __builtin_amdgcn_wmma_f32_16x16x32_bf16(false, a, false, b, (short)0, c,
                                                 false, false);
}

// ---- Tensor Data Mover: 1-D contiguous copy global -> LDS (ISA ch.8 D#) ----
// nelems 2-byte elements, must fit tile_dim0 (16 bits). Wave-uniform args.
// 6-arg builtin form (this toolchain).
__device__ __forceinline__ void tdm_load_1d(uint32_t lds_byte, const void* gptr,
                                            uint32_t nelems) {
  uint64_t ga = (uint64_t)(uintptr_t)gptr;
  v4u g0;
  g0[0] = 1u;                                           // count=1 (user D#)
  g0[1] = lds_byte;                                     // lds_addr
  g0[2] = (uint32_t)ga;                                 // global_addr[31:0]
  g0[3] = (uint32_t)((ga >> 32) & 0x01FFFFFFu) | (2u << 30); // addr[56:32]|type=2
  v8i g1;
  g1[0] = (int)(1u << 16);                              // data_size=1 (2 bytes)
  g1[1] = (int)((nelems & 0xFFFFu) << 16);              // tensor_dim0[15:0]
  g1[2] = (int)(((nelems >> 16) & 0xFFFFu) | (1u << 16)); // dim0[31:16]|dim1=1
  g1[3] = (int)((nelems & 0xFFFFu) << 16);              // tile_dim0 = nelems
  g1[4] = 1;                                            // tile_dim1=1, tile_dim2=0
  g1[5] = (int)nelems;                                  // tensor_dim0_stride lo
  g1[6] = 0;
  g1[7] = 0;
  v4i z4 = {0, 0, 0, 0};
  v8i z8 = {0, 0, 0, 0, 0, 0, 0, 0};
  __builtin_amdgcn_tensor_load_to_lds(g0, g1, z4, z4, z8, 0);
}

__device__ __forceinline__ uint32_t lds_off(const void* p) {
  // LDS aperture: flat addr[31:0] is the LDS byte offset (ISA 10.2)
  return (uint32_t)(uintptr_t)p;
}

// ---------------------------------------------------------------- cast kernel
__global__ void cast_f32_bf16_kernel(const float* __restrict__ src,
                                     unsigned short* __restrict__ dst, int n) {
  int i = blockIdx.x * blockDim.x + threadIdx.x;
  int stride = gridDim.x * blockDim.x;
  for (; i < n; i += stride) dst[i] = f2bf(src[i]);
}

// ------------------------------------------------------------ QKV projection
// grid: (M/64, 3*H). Each block: 64 rows x 64 cols (HD) for one (mat, head).
__global__ __launch_bounds__(128) void qkv_gemm_kernel(
    const unsigned short* __restrict__ x_bf,   // [4096,1024]
    const unsigned short* __restrict__ w_bf,   // [3][H][1024][64]
    const float* __restrict__ bq, const float* __restrict__ bk,
    const float* __restrict__ bv,
    unsigned short* __restrict__ qkv_bf)       // [3][B][H][S][HD]
{
  __shared__ __align__(16) unsigned short As[64 * 32];
  __shared__ __align__(16) unsigned short Bst[64 * 32];   // transposed: [n][kk]
  const int tid = threadIdx.x, lane = tid & 31, wv = tid >> 5;
  const int m0 = blockIdx.x * 64;
  const int mat = blockIdx.y >> 4, h = blockIdx.y & 15;
  const unsigned short* W = w_bf + ((size_t)mat * HH + h) * (size_t)DDIM * HD;
  const float* bias = (mat == 0) ? bq : ((mat == 1) ? bk : bv);

  v8f acc[4];
#pragma unroll
  for (int c = 0; c < 4; c++) acc[c] = (v8f){0.f, 0.f, 0.f, 0.f, 0.f, 0.f, 0.f, 0.f};

  for (int kk0 = 0; kk0 < DDIM; kk0 += 32) {
    __syncthreads();
#pragma unroll
    for (int it = 0; it < 2; it++) {
      int i = tid + it * 128;
      // A tile 64x32 row-major (4 b128 segs / row)
      {
        int row = i >> 2, seg = i & 3;
        ((uint4*)As)[i] =
            ((const uint4*)(x_bf + (size_t)(m0 + row) * DDIM + kk0))[seg];
      }
      // B tile 32x64 -> transposed into Bst[n][kk]
      {
        int row = i >> 3, n8 = (i & 7) * 8;
        uint4 d = ((const uint4*)(W + (size_t)(kk0 + row) * HD))[i & 7];
        const unsigned short* hs = (const unsigned short*)&d;
#pragma unroll
        for (int q = 0; q < 8; q++) Bst[(n8 + q) * 32 + row] = hs[q];
      }
    }
    __syncthreads();

    BfFrag af;
#pragma unroll
    for (int e = 0; e < 16; e++)
      af.u[e] = As[(wv * 16 + (lane & 15)) * 32 + aKidx(e, lane)];
#pragma unroll
    for (int c = 0; c < 4; c++) {
      BfFrag bfr;
#pragma unroll
      for (int e = 0; e < 16; e++)
        bfr.u[e] = Bst[(c * 16 + (lane & 15)) * 32 + bKidx(e, lane)];
      acc[c] = wmma_bf16(af.v, bfr.v, acc[c]);
    }
  }

  const int rowoff = (lane & 16) ? 8 : 0;
#pragma unroll
  for (int c = 0; c < 4; c++) {
    int n = c * 16 + (lane & 15);
    float bval = bias[h * HD + n];
#pragma unroll
    for (int r = 0; r < 8; r++) {
      int m = m0 + wv * 16 + r + rowoff;
      int b = m >> 11, s = m & 2047;
      size_t o = ((size_t)mat * BB * HH * SS + ((size_t)b * HH + h) * SS + s) * HD + n;
      qkv_bf[o] = f2bf(acc[c][r] + bval);
    }
  }
}

// ------------------------------------------------------- flash-attention core
// grid: (S/64, B*H). block=128 (4 waves); wave owns a 16-row query tile.
// K/V tiles streamed by the Tensor Data Mover, double-buffered so the DMA for
// block j+1 overlaps compute on block j (TENSORcnt completes in-order, so
// s_wait_tensorcnt(2) = "two oldest done" = block j's K and V tiles landed).
__global__ __launch_bounds__(128) void attn_kernel(
    const unsigned short* __restrict__ qkv_bf,
    const int* __restrict__ mask,
    unsigned short* __restrict__ attn_bf)      // [B][S][H*HD]
{
  __shared__ __align__(16) unsigned short kT[2][32 * 64];  // [key][hd] row-major
  __shared__ __align__(16) unsigned short vT[2][32 * 64];  // [key][hd] row-major
  __shared__ __align__(16) unsigned short vTt[64 * 32];    // transposed: [n][key]
  __shared__ __align__(16) unsigned short pS[4 * 16 * 32]; // per-wave P scratch
  const int tid = threadIdx.x, lane = tid & 31, wv = tid >> 5;
  const int bh = blockIdx.y;
  const int b = bh >> 4, h = bh & 15;
  const int q0 = blockIdx.x * 64 + wv * 16;
  const size_t planes = (size_t)BB * HH * SS * HD;
  const unsigned short* Qb = qkv_bf + 0 * planes + (size_t)bh * SS * HD;
  const unsigned short* Kb = qkv_bf + 1 * planes + (size_t)bh * SS * HD;
  const unsigned short* Vb = qkv_bf + 2 * planes + (size_t)bh * SS * HD;

  // Q fragments stay in registers for the whole pass (HD=64 -> 2 K-chunks)
  BfFrag aQ[2];
#pragma unroll
  for (int ch = 0; ch < 2; ch++)
#pragma unroll
    for (int e = 0; e < 16; e++)
      aQ[ch].u[e] = Qb[(size_t)(q0 + (lane & 15)) * HD + ch * 32 + aKidx(e, lane)];

  float mrow[8], lrow[8];
  v8f acc[4];
#pragma unroll
  for (int r = 0; r < 8; r++) { mrow[r] = -3.0e38f; lrow[r] = 0.f; }
#pragma unroll
  for (int c = 0; c < 4; c++) acc[c] = (v8f){0.f, 0.f, 0.f, 0.f, 0.f, 0.f, 0.f, 0.f};

  const float scale = 0.125f;  // HD^-0.5
  const int NB = SS / 32;

  // prologue: prefetch block 0 into buffer 0
  if (tid < 32) {
    tdm_load_1d(lds_off(&kT[0][0]), Kb, 32 * HD);
    tdm_load_1d(lds_off(&vT[0][0]), Vb, 32 * HD);
  }

  for (int j = 0; j < NB; j++) {
    const int cur = j & 1;
    __syncthreads();   // all waves done reading buffer cur^1 (iteration j-1)
    if (tid < 32) {    // wave 0 drives the TDM (EXEC ignored; one DMA per wave)
      if (j + 1 < NB) {
        tdm_load_1d(lds_off(&kT[cur ^ 1][0]), Kb + (size_t)(j + 1) * 32 * HD, 32 * HD);
        tdm_load_1d(lds_off(&vT[cur ^ 1][0]), Vb + (size_t)(j + 1) * 32 * HD, 32 * HD);
        __builtin_amdgcn_s_wait_tensorcnt(2);  // block j's two tiles complete
      } else {
        __builtin_amdgcn_s_wait_tensorcnt(0);
      }
    }
    __syncthreads();   // kT[cur]/vT[cur] valid for all waves

    // transpose V tile into vTt[n][key] so PV B-fragments read contiguously
#pragma unroll
    for (int it = 0; it < 2; it++) {
      int i = tid + it * 128;
      int rowk = i >> 3, n8 = (i & 7) * 8;
      uint4 d = ((const uint4*)vT[cur])[i];
      const unsigned short* hs = (const unsigned short*)&d;
#pragma unroll
      for (int q = 0; q < 8; q++) vTt[(n8 + q) * 32 + rowk] = hs[q];
    }
    __syncthreads();   // vTt valid

    // scores: two 16x16 tiles covering 32 keys (B = K^T -> read K row-major)
    v8f sc[2];
#pragma unroll
    for (int sub = 0; sub < 2; sub++) {
      v8f c = (v8f){0.f, 0.f, 0.f, 0.f, 0.f, 0.f, 0.f, 0.f};
#pragma unroll
      for (int ch = 0; ch < 2; ch++) {
        BfFrag bkf;
#pragma unroll
        for (int e = 0; e < 16; e++)
          bkf.u[e] =
              kT[cur][(sub * 16 + (lane & 15)) * 64 + ch * 32 + bKidx(e, lane)];
        c = wmma_bf16(aQ[ch].v, bkf.v, c);
      }
      sc[sub] = c;
    }

    // key-padding mask + scale
#pragma unroll
    for (int sub = 0; sub < 2; sub++) {
      int col = j * 32 + sub * 16 + (lane & 15);
      int mv = mask[b * SS + col];
#pragma unroll
      for (int r = 0; r < 8; r++)
        sc[sub][r] = mv ? sc[sub][r] * scale : -1.0e9f;
    }

    // online softmax: row reductions across the 16-lane half (wave32 shfl_xor)
    float p0v[8], p1v[8];
#pragma unroll
    for (int r = 0; r < 8; r++) {
      float t = fmaxf(sc[0][r], sc[1][r]);
      t = fmaxf(t, __shfl_xor(t, 1, 32));
      t = fmaxf(t, __shfl_xor(t, 2, 32));
      t = fmaxf(t, __shfl_xor(t, 4, 32));
      t = fmaxf(t, __shfl_xor(t, 8, 32));
      float mnew = fmaxf(mrow[r], t);
      float p0 = __expf(sc[0][r] - mnew);
      float p1 = __expf(sc[1][r] - mnew);
      p0v[r] = p0; p1v[r] = p1;
      float rs = p0 + p1;
      rs += __shfl_xor(rs, 1, 32);
      rs += __shfl_xor(rs, 2, 32);
      rs += __shfl_xor(rs, 4, 32);
      rs += __shfl_xor(rs, 8, 32);
      float alpha = __expf(mrow[r] - mnew);
      lrow[r] = lrow[r] * alpha + rs;
      mrow[r] = mnew;
#pragma unroll
      for (int c = 0; c < 4; c++) acc[c][r] *= alpha;
    }

    // P (C-layout) -> per-wave LDS scratch -> A-layout bf16 fragment
    unsigned short* myP = pS + wv * (16 * 32);
    {
      int rowoff = (lane & 16) ? 8 : 0;
#pragma unroll
      for (int r = 0; r < 8; r++) {
        myP[(r + rowoff) * 32 + (lane & 15)]      = f2bf(p0v[r]);
        myP[(r + rowoff) * 32 + 16 + (lane & 15)] = f2bf(p1v[r]);
      }
    }
    // CDNA5 split counter: same-wave LDS stores visible before reads
    asm volatile("s_wait_dscnt 0" ::: "memory");
    BfFrag aP;
#pragma unroll
    for (int e = 0; e < 16; e++)
      aP.u[e] = myP[(lane & 15) * 32 + aKidx(e, lane)];

    // O += P x V  (4 N-chunks over HD), B-fragments contiguous from vTt
#pragma unroll
    for (int c = 0; c < 4; c++) {
      BfFrag bvf;
#pragma unroll
      for (int e = 0; e < 16; e++)
        bvf.u[e] = vTt[(c * 16 + (lane & 15)) * 32 + bKidx(e, lane)];
      acc[c] = wmma_bf16(aP.v, bvf.v, acc[c]);
    }
  }

  // normalize and emit concat-head layout [B,S,H*HD] as bf16
  const int rowoff = (lane & 16) ? 8 : 0;
#pragma unroll
  for (int r = 0; r < 8; r++) {
    float inv = 1.0f / lrow[r];
    int s = q0 + r + rowoff;
#pragma unroll
    for (int c = 0; c < 4; c++)
      attn_bf[((size_t)b * SS + s) * DDIM + h * HD + c * 16 + (lane & 15)] =
          f2bf(acc[c][r] * inv);
  }
}

// ---------------------------------------------------------- output projection
// grid: (M/64, N/64). f32 epilogue + bias straight into d_out.
__global__ __launch_bounds__(128) void out_gemm_kernel(
    const unsigned short* __restrict__ A,    // [4096,1024] bf16
    const unsigned short* __restrict__ Bw,   // [1024,1024] bf16
    const float* __restrict__ bias, float* __restrict__ out)
{
  __shared__ __align__(16) unsigned short As[64 * 32];
  __shared__ __align__(16) unsigned short Bst[64 * 32];   // transposed: [n][kk]
  const int tid = threadIdx.x, lane = tid & 31, wv = tid >> 5;
  const int m0 = blockIdx.x * 64, n0 = blockIdx.y * 64;

  v8f acc[4];
#pragma unroll
  for (int c = 0; c < 4; c++) acc[c] = (v8f){0.f, 0.f, 0.f, 0.f, 0.f, 0.f, 0.f, 0.f};

  for (int kk0 = 0; kk0 < DDIM; kk0 += 32) {
    __syncthreads();
#pragma unroll
    for (int it = 0; it < 2; it++) {
      int i = tid + it * 128;
      {
        int row = i >> 2, seg = i & 3;
        ((uint4*)As)[i] =
            ((const uint4*)(A + (size_t)(m0 + row) * DDIM + kk0))[seg];
      }
      {
        int row = i >> 3, n8 = (i & 7) * 8;
        uint4 d = ((const uint4*)(Bw + (size_t)(kk0 + row) * DDIM + n0))[i & 7];
        const unsigned short* hs = (const unsigned short*)&d;
#pragma unroll
        for (int q = 0; q < 8; q++) Bst[(n8 + q) * 32 + row] = hs[q];
      }
    }
    __syncthreads();

    BfFrag af;
#pragma unroll
    for (int e = 0; e < 16; e++)
      af.u[e] = As[(wv * 16 + (lane & 15)) * 32 + aKidx(e, lane)];
#pragma unroll
    for (int c = 0; c < 4; c++) {
      BfFrag bfr;
#pragma unroll
      for (int e = 0; e < 16; e++)
        bfr.u[e] = Bst[(c * 16 + (lane & 15)) * 32 + bKidx(e, lane)];
      acc[c] = wmma_bf16(af.v, bfr.v, acc[c]);
    }
  }

  const int rowoff = (lane & 16) ? 8 : 0;
#pragma unroll
  for (int c = 0; c < 4; c++) {
    int n = n0 + c * 16 + (lane & 15);
    float bval = bias[n];
#pragma unroll
    for (int r = 0; r < 8; r++) {
      int m = m0 + wv * 16 + r + rowoff;
      out[(size_t)m * DDIM + n] = acc[c][r] + bval;
    }
  }
}

// -------------------------------------------------------------------- launch
extern "C" void kernel_launch(void* const* d_in, const int* in_sizes, int n_in,
                              void* d_out, int out_size, void* d_ws, size_t ws_size,
                              hipStream_t stream) {
  (void)in_sizes; (void)n_in; (void)out_size; (void)ws_size;
  const float* x   = (const float*)d_in[0];
  const int*   msk = (const int*)d_in[1];
  const float* Wq  = (const float*)d_in[2];
  const float* bq  = (const float*)d_in[3];
  const float* Wk  = (const float*)d_in[4];
  const float* bk  = (const float*)d_in[5];
  const float* Wv  = (const float*)d_in[6];
  const float* bv  = (const float*)d_in[7];
  const float* Wo  = (const float*)d_in[8];
  const float* bo  = (const float*)d_in[9];

  // workspace layout (bf16 elements), ~48 MB total
  unsigned short* ws      = (unsigned short*)d_ws;
  unsigned short* x_bf    = ws;                                     // 4M elems
  unsigned short* w_bf    = x_bf + (size_t)BB * SS * DDIM;          // 3M elems
  unsigned short* wo_bf   = w_bf + (size_t)3 * HH * DDIM * HD;      // 1M elems
  unsigned short* qkv_bf  = wo_bf + (size_t)DDIM * DDIM;            // 12M elems
  unsigned short* attn_bf = qkv_bf + (size_t)3 * BB * HH * SS * HD; // 4M elems

  const int NX = BB * SS * DDIM;       // 4194304
  const int NW = HH * DDIM * HD;       // 1048576
  cast_f32_bf16_kernel<<<2048, 256, 0, stream>>>(x,  x_bf, NX);
  cast_f32_bf16_kernel<<<1024, 256, 0, stream>>>(Wq, w_bf, NW);
  cast_f32_bf16_kernel<<<1024, 256, 0, stream>>>(Wk, w_bf + (size_t)NW, NW);
  cast_f32_bf16_kernel<<<1024, 256, 0, stream>>>(Wv, w_bf + (size_t)2 * NW, NW);
  cast_f32_bf16_kernel<<<1024, 256, 0, stream>>>(Wo, wo_bf, NW);

  qkv_gemm_kernel<<<dim3(BB * SS / 64, 3 * HH), 128, 0, stream>>>(
      x_bf, w_bf, bq, bk, bv, qkv_bf);

  attn_kernel<<<dim3(SS / 64, BB * HH), 128, 0, stream>>>(qkv_bf, msk, attn_bf);

  out_gemm_kernel<<<dim3(BB * SS / 64, DDIM / 64), 128, 0, stream>>>(
      attn_bf, wo_bf, bo, (float*)d_out);
}
